// ViGBlock_34668976013756
// MI455X (gfx1250) — compile-verified, matrix operand order checked by reading, not run on hardware
//
#include <hip/hip_runtime.h>
#include <math.h>

// ViG block for MI455X (gfx1250, wave32, WMMA).
// B=8, N=3136, C=192, K=9, M=25088.  All GEMM-shaped math (sim + 9 GEMMs,
// ~71 GFLOP total) runs on v_wmma_f32_16x16x32_f16 with f32 accumulate.
// x and all weights are pre-converted to f16 once so the hot loops are pure
// global_load_b128 -> v_wmma chains.  sim (315MB) is never materialized:
// top-9 selection is fused into the WMMA similarity kernel.  The GEMM's f16
// A-staging uses async global->LDS copies (ASYNCcnt) via the gfx1250
// builtin (probe-confirmed present; params are int4* pointers).

#define BSZ   8
#define NPTS  3136
#define CDIM  192
#define KNN   9
#define MROWS (BSZ * NPTS)     // 25088
#define NTILE (NPTS / 16)      // 196
#define NRB   (MROWS / 128)    // 196 row-blocks for GEMM / stats

#if defined(__has_builtin)
#  if __has_builtin(__builtin_amdgcn_global_load_async_to_lds_b128)
#    define USE_ASYNC_LDS 1
#  endif
#endif
#ifndef USE_ASYNC_LDS
#  define USE_ASYNC_LDS 0
#endif

typedef __attribute__((ext_vector_type(16))) _Float16 v16h;
typedef __attribute__((ext_vector_type(8)))  _Float16 v8h;
typedef __attribute__((ext_vector_type(4)))  _Float16 v4h;
typedef __attribute__((ext_vector_type(2)))  _Float16 v2h;
typedef __attribute__((ext_vector_type(8)))  float    v8f;

// Async-LDS builtin parameter type (from the toolchain diagnostic):
// pointer to int __attribute__((vector_size(16))).
typedef int v4i_t __attribute__((vector_size(16)));
typedef __attribute__((address_space(1))) v4i_t* as1_v4i_p;
typedef __attribute__((address_space(3))) v4i_t* as3_v4i_p;

__device__ __forceinline__ float gelu_f(float x) {
  // exact GELU: 0.5*x*(1+erf(x/sqrt(2)))
  return 0.5f * x * (1.0f + erff(x * 0.70710678118654752440f));
}

// 16-half WMMA operand per the gfx1250 16-bit A/B VGPR layout:
// elements 0..7 = K {koff..koff+7}, elements 8..15 = K {16+koff..16+koff+7}.
// p already includes the lane's koff; both chunks are 16B aligned.
__device__ __forceinline__ v16h load16_h(const _Float16* p) {
  v8h lo = *(const v8h*)(p);
  v8h hi = *(const v8h*)(p + 16);
  v16h r;
  #pragma unroll
  for (int i = 0; i < 8; ++i) { r[i] = lo[i]; r[8 + i] = hi[i]; }
  return r;
}

// Branchless sorted-descending top-9 insert (fully unrolled -> registers).
__device__ __forceinline__ void topk9_insert(float v, int cg, float* tv, int* ti) {
  #pragma unroll
  for (int p = 8; p >= 1; --p) {
    bool gt = v > tv[p];
    bool up = v > tv[p - 1];
    float nv = gt ? (up ? tv[p - 1] : v)  : tv[p];
    int   ni = gt ? (up ? ti[p - 1] : cg) : ti[p];
    tv[p] = nv; ti[p] = ni;
  }
  bool g0 = v > tv[0];
  float nv0 = g0 ? v : tv[0];
  int   ni0 = g0 ? cg : ti[0];
  tv[0] = nv0; ti[0] = ni0;
}

// ---------------------------------------------------------------------------
// f32 -> f16 conversion (vectorized x4); all our sizes are multiples of 4.
// ---------------------------------------------------------------------------
__global__ __launch_bounds__(256) void f32_to_f16_kernel(const float* __restrict__ in,
                                                         _Float16* __restrict__ out,
                                                         int n4) {
  int i = blockIdx.x * 256 + threadIdx.x;
  if (i < n4) {
    float4 f = *(const float4*)&in[i * 4];
    v4h h;
    h[0] = (_Float16)f.x; h[1] = (_Float16)f.y;
    h[2] = (_Float16)f.z; h[3] = (_Float16)f.w;
    *(v4h*)&out[i * 4] = h;
  }
}

// ---------------------------------------------------------------------------
// Kernel 1: fused similarity (WMMA) + per-row top-9 neighbor selection.
// Block = 256 thr (8 waves) owns 16 rows of one batch. Each wave sweeps
// column tiles jt = w, w+8, ...: 12 b128 loads + 6 chained WMMAs per tile.
// ---------------------------------------------------------------------------
__global__ __launch_bounds__(256) void topk_kernel(const _Float16* __restrict__ xh,
                                                   int* __restrict__ graph) {
  __shared__ float simt[8 * 256];     // per-wave 16x16 f32 tile (transpose)
  __shared__ float lv[8 * 32 * 9];    // per-lane top-9 values
  __shared__ int   li[8 * 32 * 9];    // per-lane top-9 indices

  const int b  = blockIdx.x / NTILE;
  const int n0 = (blockIdx.x % NTILE) * 16;
  const int t  = threadIdx.x;
  const _Float16* xb = xh + (size_t)b * NPTS * CDIM;

  const int w = t >> 5, lane = t & 31, l16 = lane & 15, half = lane >> 4;
  const int koff = half * 8;

  // A operands (the block's 16 rows) straight from global f16, kept in
  // registers for the whole sweep (reused ~25x per wave).
  v16h a[6];
  #pragma unroll
  for (int ks = 0; ks < 6; ++ks)
    a[ks] = load16_h(xb + (size_t)(n0 + l16) * CDIM + ks * 32 + koff);

  float tv[9]; int ti[9];
  #pragma unroll
  for (int q = 0; q < 9; ++q) { tv[q] = -__builtin_inff(); ti[q] = 0; }

  float* mys = &simt[w * 256];
  for (int jt = w; jt < NTILE; jt += 8) {
    // Preload all 6 B operands, then run the hazard-free D->C WMMA chain.
    v16h bb[6];
    #pragma unroll
    for (int ks = 0; ks < 6; ++ks)
      bb[ks] = load16_h(xb + (size_t)(jt * 16 + l16) * CDIM + ks * 32 + koff);
    v8f acc;
    #pragma unroll
    for (int e = 0; e < 8; ++e) acc[e] = 0.0f;
    #pragma unroll
    for (int ks = 0; ks < 6; ++ks)
      acc = __builtin_amdgcn_wmma_f32_16x16x32_f16(false, a[ks], false, bb[ks],
                                                   (short)0, acc, false, false);
    // D layout: elem v -> row M = half*8+v, col N = l16. Transpose via LDS
    // (per-wave region; same-wave DS ops are in-order).
    #pragma unroll
    for (int v = 0; v < 8; ++v) mys[(half * 8 + v) * 16 + l16] = acc[v];
    // Lane (l16, half) scans row l16, columns koff..koff+7.
    #pragma unroll
    for (int q = 0; q < 8; ++q) {
      float v = mys[l16 * 16 + koff + q];
      if (v > tv[8]) topk9_insert(v, jt * 16 + koff + q, tv, ti);
    }
  }

  const int lb = (w * 32 + lane) * 9;
  #pragma unroll
  for (int q = 0; q < 9; ++q) { lv[lb + q] = tv[q]; li[lb + q] = ti[q]; }
  __syncthreads();

  // Merge: row r is covered by 16 sorted lists (8 waves x 2 column-halves).
  if (t < 16) {
    float mv[9]; int mi[9];
    #pragma unroll
    for (int q = 0; q < 9; ++q) { mv[q] = -__builtin_inff(); mi[q] = 0; }
    for (int src = 0; src < 16; ++src) {
      int listlane = (src & 7) * 32 + ((src >> 3) << 4) + t;
      int base = listlane * 9;
      #pragma unroll
      for (int q = 0; q < 9; ++q) {
        float v = lv[base + q];
        if (!(v > mv[8])) break;          // source list is sorted descending
        topk9_insert(v, li[base + q], mv, mi);
      }
    }
    #pragma unroll
    for (int q = 0; q < 9; ++q)
      graph[((size_t)b * NPTS + n0 + t) * KNN + q] = mi[q];
  }
}

// ---------------------------------------------------------------------------
// Kernel 2: generic WMMA GEMM  Y = preop(A) @ W^T + bias [+res1][+res2] [gelu]
//   A sources: Ah (f16, used when preop==0; staged via async global->LDS
//   copies) or A (f32, with fused batchnorm+GELU pre-op during staging).
//   W is pre-converted f16.  Optional dual store of Y as f16 (Yh) for
//   consumers that use it as a later GEMM's A operand.
// Block = 256 thr, tile 128 rows x 64 cols, K-step 32.
// ---------------------------------------------------------------------------
__global__ __launch_bounds__(256) void gemm_kernel(
    const float* __restrict__ A, const _Float16* __restrict__ Ah,
    const _Float16* __restrict__ Wh,
    const float* __restrict__ bias, const float* __restrict__ res1,
    const float* __restrict__ res2, float* __restrict__ Y,
    _Float16* __restrict__ Yh,
    int Cin, int Cout,
    const float* __restrict__ mu, const float* __restrict__ rstd,
    const float* __restrict__ gamma, const float* __restrict__ beta,
    int preop, int gelu_out) {
  __shared__ _Float16 as16[128 * 40];    // 128 rows x 32 halves (pad->40)

  const int t  = threadIdx.x;
  const int r0 = blockIdx.x * 128;
  const int jb = blockIdx.y * 64;
  const int w = t >> 5, lane = t & 31, l16 = lane & 15, half = lane >> 4;
  const int koff = half * 8;

  v8f acc[4];
  #pragma unroll
  for (int j = 0; j < 4; ++j)
    #pragma unroll
    for (int e = 0; e < 8; ++e) acc[j][e] = 0.0f;

  const int nks = Cin >> 5;
  const int lr = t >> 3;            // f32 staging row 0..31
  const int c4 = (t & 7) * 4;       // f32 staging col 0..28

  for (int ks = 0; ks < nks; ++ks) {
    __syncthreads();
    if (Ah) {
      // f16 A: tile = 128 rows x 32 halves = 512 chunks of 8 halves (16B).
      // Each thread copies chunks t and t+256 (consecutive threads ->
      // consecutive 16B -> coalesced; all offsets 16B aligned).
      #pragma unroll
      for (int u = 0; u < 2; ++u) {
        const int cidx  = t + u * 256;
        const int row_l = cidx >> 2;
        const int hc    = (cidx & 3) * 8;
        const _Float16* gp = &Ah[(size_t)(r0 + row_l) * Cin + ks * 32 + hc];
        _Float16* lp = &as16[row_l * 40 + hc];
#if USE_ASYNC_LDS
        __builtin_amdgcn_global_load_async_to_lds_b128(
            (as1_v4i_p)gp, (as3_v4i_p)lp, 0, 0);
#else
        *(v8h*)lp = *(const v8h*)gp;
#endif
      }
    } else {
      // f32 A with fused batchnorm+GELU, converted to f16 during staging.
      #pragma unroll
      for (int p = 0; p < 4; ++p) {
        int row_l = p * 32 + lr;
        const float4 v4 = *(const float4*)&A[(size_t)(r0 + row_l) * Cin + ks * 32 + c4];
        float vv[4] = {v4.x, v4.y, v4.z, v4.w};
        v4h hv;
        #pragma unroll
        for (int i = 0; i < 4; ++i) {
          float f = vv[i];
          if (preop == 2) {
            int c = ks * 32 + c4 + i;
            f = (f - mu[c]) * rstd[c] * gamma[c] + beta[c];
            f = gelu_f(f);
          } else if (preop == 1) {
            f = gelu_f(f);
          }
          hv[i] = (_Float16)f;
        }
        *(v4h*)&as16[row_l * 40 + c4] = hv;   // single ds_store_b64
        if (ks + 1 < nks)
          __builtin_prefetch(&A[(size_t)(r0 + row_l) * Cin + (ks + 1) * 32 + c4], 0, 0);
      }
    }
#if USE_ASYNC_LDS
    if (Ah) {
#  if __has_builtin(__builtin_amdgcn_s_wait_asynccnt)
      __builtin_amdgcn_s_wait_asynccnt(0);
#  else
      asm volatile("s_wait_asynccnt 0" ::: "memory");
#  endif
    }
#endif
    __syncthreads();

    v16h av = load16_h(&as16[(w * 16 + l16) * 40 + koff]);
    #pragma unroll
    for (int j = 0; j < 4; ++j) {
      // B tile: lane = output column n, reads Wh row (jb + j*16 + l16).
      v16h bv = load16_h(&Wh[(size_t)(jb + j * 16 + l16) * Cin + ks * 32 + koff]);
      acc[j] = __builtin_amdgcn_wmma_f32_16x16x32_f16(false, av, false, bv,
                                                      (short)0, acc[j], false, false);
    }
  }

  // Epilogue: bias + residual(s) + optional GELU + optional dual f16 store.
  #pragma unroll
  for (int j = 0; j < 4; ++j) {
    int col = jb + j * 16 + l16;
    float bsv = bias[col];
    #pragma unroll
    for (int v = 0; v < 8; ++v) {
      int row = r0 + w * 16 + half * 8 + v;
      size_t o = (size_t)row * Cout + col;
      float f = acc[j][v] + bsv;
      if (res1) f += res1[o];
      if (res2) f += res2[o];
      if (gelu_out) f = gelu_f(f);
      Y[o] = f;
      if (Yh) Yh[o] = (_Float16)f;
    }
  }
}

// ---------------------------------------------------------------------------
// Column stats (mean/var over the M row dimension), deterministic:
// fixed-order per-block partials, then a fixed-order reduction.
// ---------------------------------------------------------------------------
__global__ __launch_bounds__(256) void colstats_partial(const float* __restrict__ Yp,
                                                        float* __restrict__ part,
                                                        int Cout) {
  const int r0 = blockIdx.x * 128;
  const int t  = threadIdx.x;
  float s[3], q[3];
  #pragma unroll
  for (int i = 0; i < 3; ++i) { s[i] = 0.0f; q[i] = 0.0f; }
  for (int r = 0; r < 128; ++r) {
    const float* rowp = &Yp[(size_t)(r0 + r) * Cout];
    #pragma unroll
    for (int i = 0; i < 3; ++i) {
      int c = t + i * 256;
      if (c < Cout) { float v = rowp[c]; s[i] += v; q[i] += v * v; }
    }
  }
  #pragma unroll
  for (int i = 0; i < 3; ++i) {
    int c = t + i * 256;
    if (c < Cout) {
      part[(size_t)blockIdx.x * 2 * Cout + c]        = s[i];
      part[(size_t)blockIdx.x * 2 * Cout + Cout + c] = q[i];
    }
  }
}

__global__ __launch_bounds__(256) void colstats_finalize(const float* __restrict__ part,
                                                         float* __restrict__ muv,
                                                         float* __restrict__ rsv,
                                                         int Cout) {
  for (int c = threadIdx.x; c < Cout; c += 256) {
    float s = 0.0f, q = 0.0f;
    for (int b = 0; b < NRB; ++b) {
      s += part[(size_t)b * 2 * Cout + c];
      q += part[(size_t)b * 2 * Cout + Cout + c];
    }
    float m   = s / (float)MROWS;
    float var = q / (float)MROWS - m * m;
    muv[c] = m;
    rsv[c] = rsqrtf(var + 1e-5f);
  }
}

// ---------------------------------------------------------------------------
// Max-relative graph aggregation + interleaved [h, agg] stack, stored
// directly as f16 (it is only ever consumed as the fc GEMM's A operand):
//   Sh[n, 2c] = h[n,c];  Sh[n, 2c+1] = max_k h[idx_k, c] - h[n, c]
// ---------------------------------------------------------------------------
__global__ __launch_bounds__(CDIM) void gather_stack_kernel(const float* __restrict__ H,
                                                            const int* __restrict__ graph,
                                                            _Float16* __restrict__ Sh) {
  const int row = blockIdx.x;          // 0..MROWS-1
  const int b   = row / NPTS;
  const int c   = threadIdx.x;
  float hv = H[(size_t)row * CDIM + c];
  float mx = -__builtin_inff();
  #pragma unroll
  for (int k = 0; k < KNN; ++k) {
    int g = graph[(size_t)row * KNN + k];
    mx = fmaxf(mx, H[((size_t)b * NPTS + g) * CDIM + c]);
  }
  v2h o;
  o[0] = (_Float16)hv;
  o[1] = (_Float16)(mx - hv);
  *(v2h*)&Sh[(size_t)row * 2 * CDIM + 2 * c] = o;
}

// Elementwise exact GELU with dual f32 + f16 store.
__global__ __launch_bounds__(256) void gelu_kernel(const float* __restrict__ in,
                                                   float* __restrict__ out,
                                                   _Float16* __restrict__ outh, int n) {
  int i = blockIdx.x * blockDim.x + threadIdx.x;
  if (i < n) {
    float f = gelu_f(in[i]);
    out[i] = f;
    outh[i] = (_Float16)f;
  }
}

// ---------------------------------------------------------------------------
// Host orchestration.
// Input order (flattened setup_inputs dict):
//  0:x  1-6:p_in1{W1,b1,g,beta,W2,b2}  7-12:p_out1  13-18:p_in2  19-24:p_out2
//  25:fc_W  26:fc_b
// ---------------------------------------------------------------------------
extern "C" void kernel_launch(void* const* d_in, const int* in_sizes, int n_in,
                              void* d_out, int out_size, void* d_ws, size_t ws_size,
                              hipStream_t stream) {
  (void)in_sizes; (void)n_in; (void)out_size; (void)ws_size;
  const float* x    = (const float*)d_in[0];
  const float* i1W1 = (const float*)d_in[1];
  const float* i1b1 = (const float*)d_in[2];
  const float* i1g  = (const float*)d_in[3];
  const float* i1be = (const float*)d_in[4];
  const float* i1W2 = (const float*)d_in[5];
  const float* i1b2 = (const float*)d_in[6];
  const float* o1W1 = (const float*)d_in[7];
  const float* o1b1 = (const float*)d_in[8];
  const float* o1g  = (const float*)d_in[9];
  const float* o1be = (const float*)d_in[10];
  const float* o1W2 = (const float*)d_in[11];
  const float* o1b2 = (const float*)d_in[12];
  const float* i2W1 = (const float*)d_in[13];
  const float* i2b1 = (const float*)d_in[14];
  const float* i2g  = (const float*)d_in[15];
  const float* i2be = (const float*)d_in[16];
  const float* i2W2 = (const float*)d_in[17];
  const float* i2b2 = (const float*)d_in[18];
  const float* o2W1 = (const float*)d_in[19];
  const float* o2b1 = (const float*)d_in[20];
  const float* o2g  = (const float*)d_in[21];
  const float* o2be = (const float*)d_in[22];
  const float* o2W2 = (const float*)d_in[23];
  const float* o2b2 = (const float*)d_in[24];
  const float* fcW  = (const float*)d_in[25];
  const float* fcb  = (const float*)d_in[26];

  char* ws = (char*)d_ws;
  size_t off = 0;
  auto take = [&](size_t bytes) -> char* {
    char* p = ws + off;
    off = (off + bytes + 255) & ~(size_t)255;
    return p;
  };
  int*       graph = (int*)      take((size_t)MROWS * KNN * 4);
  float*     T     = (float*)    take((size_t)MROWS * 768 * 4);  // gemm1 outputs
  float*     H1    = (float*)    take((size_t)MROWS * CDIM * 4); // also H2
  float*     HFC   = (float*)    take((size_t)MROWS * CDIM * 4); // also G
  float*     X2    = (float*)    take((size_t)MROWS * CDIM * 4);
  float*     part  = (float*)    take((size_t)NRB * 2 * 768 * 4);
  float*     muv   = (float*)    take(768 * 4);
  float*     rsv   = (float*)    take(768 * 4);
  _Float16*  xhh   = (_Float16*) take((size_t)MROWS * CDIM * 2);
  _Float16*  Sh    = (_Float16*) take((size_t)MROWS * 2 * CDIM * 2);
  _Float16*  HFCh  = (_Float16*) take((size_t)MROWS * CDIM * 2);
  _Float16*  X2h   = (_Float16*) take((size_t)MROWS * CDIM * 2);
  _Float16*  Gh    = (_Float16*) take((size_t)MROWS * CDIM * 2);
  _Float16*  i1W1h = (_Float16*) take((size_t)CDIM * CDIM * 2);
  _Float16*  i1W2h = (_Float16*) take((size_t)CDIM * CDIM * 2);
  _Float16*  o1W1h = (_Float16*) take((size_t)CDIM * CDIM * 2);
  _Float16*  o1W2h = (_Float16*) take((size_t)CDIM * CDIM * 2);
  _Float16*  fcWh  = (_Float16*) take((size_t)CDIM * 2 * CDIM * 2);
  _Float16*  i2W1h = (_Float16*) take((size_t)4 * CDIM * CDIM * 2);
  _Float16*  i2W2h = (_Float16*) take((size_t)4 * CDIM * CDIM * 2);
  _Float16*  o2W1h = (_Float16*) take((size_t)4 * CDIM * CDIM * 2);
  _Float16*  o2W2h = (_Float16*) take((size_t)4 * CDIM * CDIM * 2);

  const dim3 blk(256);
  auto cvt = [&](const float* in, _Float16* out, int n) {
    int n4 = n / 4;
    f32_to_f16_kernel<<<dim3((n4 + 255) / 256), blk, 0, stream>>>(in, out, n4);
  };
  auto gemm = [&](const float* Ap, const _Float16* Ahp, const _Float16* Whp,
                  const float* bp, const float* r1, const float* r2,
                  float* Yp, _Float16* Yhp, int Cin, int Cout,
                  const float* mu_, const float* rs_, const float* g_,
                  const float* be_, int preop, int gout) {
    gemm_kernel<<<dim3(MROWS / 128, Cout / 64), blk, 0, stream>>>(
        Ap, Ahp, Whp, bp, r1, r2, Yp, Yhp, Cin, Cout, mu_, rs_, g_, be_,
        preop, gout);
  };
  auto stats = [&](const float* Yp, int Cout) {
    colstats_partial<<<dim3(NRB), blk, 0, stream>>>(Yp, part, Cout);
    colstats_finalize<<<dim3(1), blk, 0, stream>>>(part, muv, rsv, Cout);
  };

  // 0) One-time f16 conversions (x: 9.2MB, weights: ~1.6MB total).
  cvt(x, xhh, MROWS * CDIM);
  cvt(i1W1, i1W1h, CDIM * CDIM);
  cvt(i1W2, i1W2h, CDIM * CDIM);
  cvt(o1W1, o1W1h, CDIM * CDIM);
  cvt(o1W2, o1W2h, CDIM * CDIM);
  cvt(fcW,  fcWh,  CDIM * 2 * CDIM);
  cvt(i2W1, i2W1h, 4 * CDIM * CDIM);
  cvt(i2W2, i2W2h, 4 * CDIM * CDIM);
  cvt(o2W1, o2W1h, 4 * CDIM * CDIM);
  cvt(o2W2, o2W2h, 4 * CDIM * CDIM);

  // 1) kNN graph from fused sim + top-9.
  topk_kernel<<<dim3(BSZ * NTILE), blk, 0, stream>>>(xhh, graph);

  // 2) h = two_layer(x, p_in1)                  -> H1
  gemm(nullptr, xhh, i1W1h, i1b1, nullptr, nullptr, T, nullptr, CDIM, CDIM,
       nullptr, nullptr, nullptr, nullptr, 0, 0);
  stats(T, CDIM);
  gemm(T, nullptr, i1W2h, i1b2, x, nullptr, H1, nullptr, CDIM, CDIM,
       muv, rsv, i1g, i1be, 2, 0);

  // 3) graph conv: stack [h, maxrel] (f16) -> fc -> gelu  -> HFC (+f16)
  gather_stack_kernel<<<dim3(MROWS), dim3(CDIM), 0, stream>>>(H1, graph, Sh);
  gemm(nullptr, Sh, fcWh, fcb, nullptr, nullptr, HFC, HFCh, 2 * CDIM, CDIM,
       nullptr, nullptr, nullptr, nullptr, 0, 1);

  // 4) x2 = two_layer(HFC, p_out1) + x           -> X2 (+f16)
  gemm(nullptr, HFCh, o1W1h, o1b1, nullptr, nullptr, T, nullptr, CDIM, CDIM,
       nullptr, nullptr, nullptr, nullptr, 0, 0);
  stats(T, CDIM);
  gemm(T, nullptr, o1W2h, o1b2, HFC, x, X2, X2h, CDIM, CDIM,
       muv, rsv, o1g, o1be, 2, 0);

  // 5) h2 = two_layer(X2, p_in2)                 -> H1 (reuse)
  gemm(nullptr, X2h, i2W1h, i2b1, nullptr, nullptr, T, nullptr, CDIM, 4 * CDIM,
       nullptr, nullptr, nullptr, nullptr, 0, 0);
  stats(T, 4 * CDIM);
  gemm(T, nullptr, i2W2h, i2b2, X2, nullptr, H1, nullptr, 4 * CDIM, CDIM,
       muv, rsv, i2g, i2be, 2, 0);

  // 6) G = gelu(h2)                              -> HFC (reuse, +f16)
  gelu_kernel<<<dim3((MROWS * CDIM + 255) / 256), blk, 0, stream>>>(
      H1, HFC, Gh, MROWS * CDIM);

  // 7) out = two_layer(G, p_out2) + X2           -> d_out
  gemm(nullptr, Gh, o2W1h, o2b1, nullptr, nullptr, T, nullptr, CDIM, 4 * CDIM,
       nullptr, nullptr, nullptr, nullptr, 0, 0);
  stats(T, 4 * CDIM);
  gemm(T, nullptr, o2W2h, o2b2, HFC, X2, (float*)d_out, nullptr, 4 * CDIM, CDIM,
       muv, rsv, o2g, o2be, 2, 0);
}
